// DCNv2_70428873719887
// MI455X (gfx1250) — matrix operand order repaired and checked by reading, not compiled, not run on hardware
//
#include <hip/hip_runtime.h>
#include <hip/hip_bf16.h>

typedef __attribute__((ext_vector_type(16))) _Float16 v16h;
typedef __attribute__((ext_vector_type(8)))  float    v8f;

#define HWSZ 4096   // 64*64 plane
#define KTOT 1152   // 128*9

// ---------------------------------------------------------------------------
// Weight transform: weight[o][c][k] (fp32) -> Wt[o][k*128 + c] (f16)
// ---------------------------------------------------------------------------
__global__ __launch_bounds__(256) void dcn_wt_kernel(const float* __restrict__ w,
                                                     _Float16* __restrict__ wt) {
    int idx = blockIdx.x * 256 + threadIdx.x;      // 128*1152 = 147456 total
    if (idx >= 128 * KTOT) return;
    int o  = idx / KTOT;
    int kc = idx - o * KTOT;
    int k  = kc >> 7;          // kc = k*128 + c
    int c  = kc & 127;
    wt[idx] = (_Float16)w[(o * 128 + c) * 9 + k];
}

// ---------------------------------------------------------------------------
// Main implicit-GEMM DCNv2 kernel.
// Grid: 2048 blocks = B(8) * H(64) * (W/16 = 4).  Block: 256 threads (8 waves).
// Block computes 128 out-channels x 16 pixels; each wave one 16x16 D tile,
// accumulated over 36 K-chunks of 32 via v_wmma_f32_16x16x32_f16.
// B-tile double-buffered in LDS: one barrier per chunk, staging overlaps WMMA.
// ---------------------------------------------------------------------------
__global__ __launch_bounds__(256) void dcn_main_kernel(
    const float* __restrict__ x,       // [8][128][64][64]
    const float* __restrict__ offset,  // [8][18][64][64]
    const float* __restrict__ mask,    // [8][9][64][64]
    const _Float16* __restrict__ wt,   // [128][1152]
    const float* __restrict__ bias,    // [128]
    float* __restrict__ out)           // [8][128][64][64]
{
    const int blk  = blockIdx.x;
    const int wt4  = blk & 3;            // 16-pixel segment of the row
    const int ho   = (blk >> 2) & 63;
    const int b    = blk >> 8;
    const int wo0  = wt4 * 16;

    __shared__ int      s_off[9][16][4];    // clamped corner offsets (yi*64+xi)
    __shared__ float    s_wgt[9][16][4];    // bilinear weights * mask (0 if OOB)
    __shared__ _Float16 s_b[2][16][40];     // double-buffered B-tile [n][c_rel]

    const int tid = threadIdx.x;

    // ---- precompute bilinear parameters for 9 taps x 16 pixels -------------
    if (tid < 144) {
        const int k  = tid >> 4;          // 0..8
        const int n  = tid & 15;          // pixel within tile
        const int wo = wo0 + n;
        const int ky = k / 3, kx = k - 3 * ky;

        const size_t pix = (size_t)ho * 64 + wo;
        const float off_y = offset[((size_t)b * 18 + 2 * k    ) * HWSZ + pix];
        const float off_x = offset[((size_t)b * 18 + 2 * k + 1) * HWSZ + pix];
        const float m     = mask  [((size_t)b * 9  + k        ) * HWSZ + pix];

        const float sy = (float)(ky + ho - 1) + off_y;
        const float sx = (float)(kx + wo - 1) + off_x;
        const float y0f = floorf(sy), x0f = floorf(sx);
        const float wy1 = sy - y0f,  wx1 = sx - x0f;
        const int y0 = (int)y0f, x0 = (int)x0f;

        #pragma unroll
        for (int i = 0; i < 4; ++i) {
            const int yc = y0 + (i >> 1);
            const int xc = x0 + (i & 1);
            const bool valid = (yc >= 0) & (yc <= 63) & (xc >= 0) & (xc <= 63);
            const int yi = min(max(yc, 0), 63);
            const int xi = min(max(xc, 0), 63);
            const float wy = (i >> 1) ? wy1 : (1.0f - wy1);
            const float wx = (i & 1)  ? wx1 : (1.0f - wx1);
            s_off[k][n][i] = yi * 64 + xi;
            s_wgt[k][n][i] = valid ? (m * wy * wx) : 0.0f;
        }
    }
    __syncthreads();

    // batch-local plane block; all sample offsets fit in 32-bit element index
    const float* __restrict__ xb = x + (size_t)b * 128 * HWSZ;

    const int wave = tid >> 5;        // 0..7 -> M-tile (16 out channels)
    const int lane = tid & 31;
    const int o0   = wave * 16;
    const int nn   = lane & 15;       // N / M-row index within tile
    const int hi   = lane >> 4;       // lane half: K split / M+8

    // staging coordinates (uniform role per thread across all chunks)
    const int sn = tid & 15;          // pixel
    const int sc = tid >> 4;          // channel 0..15 (and +16)

    // ---- stage one chunk (32 channels x 16 pixels) into buffer `buf` ------
    auto stage = [&](int chunk, int buf) {
        const int k = chunk >> 2;
        const unsigned c0 = (unsigned)(chunk & 3) * 32u;
        const int4   of = *(const int4*)  s_off[k][sn];
        const float4 wg = *(const float4*)s_wgt[k][sn];
        const unsigned cb0 = (c0 + (unsigned)sc) << 12;   // element index base
        const unsigned cb1 = cb0 + (16u << 12);
        const float v0 = wg.x * xb[cb0 + of.x] + wg.y * xb[cb0 + of.y]
                       + wg.z * xb[cb0 + of.z] + wg.w * xb[cb0 + of.w];
        const float v1 = wg.x * xb[cb1 + of.x] + wg.y * xb[cb1 + of.y]
                       + wg.z * xb[cb1 + of.z] + wg.w * xb[cb1 + of.w];
        s_b[buf][sn][sc]      = (_Float16)v0;
        s_b[buf][sn][sc + 16] = (_Float16)v1;
    };

    // accumulator initialized with bias per C/D layout (VGPR r -> M = hi*8+r)
    v8f acc;
    #pragma unroll
    for (int r = 0; r < 8; ++r) acc[r] = bias[o0 + hi * 8 + r];

    stage(0, 0);
    __syncthreads();

    // ---- K loop: 36 chunks of 32 (chunk = fixed tap k, 32 channels) --------
    for (int cc = 0; cc < 36; ++cc) {
        const int cur = cc & 1;

        // A fragment: row o = o0+nn of Wt, K-chunk split per lane half
        const int kb = cc * 32 + (hi ? 8 : 0);
        const _Float16* arow = wt + (size_t)(o0 + nn) * KTOT + kb;
        v16h aa;
        *((float4*)&aa)     = *(const float4*)(arow);        // K +0..7
        *((float4*)&aa + 1) = *(const float4*)(arow + 16);   // K +16..23
        if (cc + 1 < 36) __builtin_prefetch(arow + 32, 0, 3);

        // B fragment from LDS (staged last iteration)
        const _Float16* brow = &s_b[cur][nn][hi ? 8 : 0];
        v16h bb;
        *((float4*)&bb)     = *(const float4*)(brow);
        *((float4*)&bb + 1) = *(const float4*)(brow + 16);

        // overlap: stage next chunk into the other buffer
        if (cc + 1 < 36) stage(cc + 1, cur ^ 1);

        acc = __builtin_amdgcn_wmma_f32_16x16x32_f16(
                  false, aa, false, bb, (short)0, acc, false, false);
        __syncthreads();
    }

    // ---- store D: VGPR r -> M = hi*8 + r, N = nn ---------------------------
    float* orow = out + (((size_t)b * 128 + o0 + hi * 8) * HWSZ)
                      + (size_t)ho * 64 + wo0 + nn;
    #pragma unroll
    for (int r = 0; r < 8; ++r) orow[(size_t)r * HWSZ] = acc[r];
}

// ---------------------------------------------------------------------------
extern "C" void kernel_launch(void* const* d_in, const int* in_sizes, int n_in,
                              void* d_out, int out_size, void* d_ws, size_t ws_size,
                              hipStream_t stream) {
    const float* x      = (const float*)d_in[0];
    const float* offset = (const float*)d_in[1];
    const float* mask   = (const float*)d_in[2];
    const float* weight = (const float*)d_in[3];
    const float* bias   = (const float*)d_in[4];
    float* out          = (float*)d_out;
    _Float16* wtb       = (_Float16*)d_ws;     // needs 128*1152*2 = 288 KB

    dcn_wt_kernel<<<576, 256, 0, stream>>>(weight, wtb);
    dcn_main_kernel<<<2048, 256, 0, stream>>>(x, offset, mask, wtb, bias, out);
}
